// Block_46746424050070
// MI455X (gfx1250) — matrix operand order repaired and checked by reading, not compiled, not run on hardware
//
#include <hip/hip_runtime.h>
#include <hip/hip_bf16.h>

// ---------------------------------------------------------------------------
// Transformer block (pre-norm, GPT-2) for MI455X / gfx1250.
// - all GEMMs on v_wmma_f32_16x16x32_bf16 (wave32, f32 accum)
// - flash attention (online softmax, DPP16 row reductions)
// - ALL global->LDS staging done by the Tensor Data Mover
//   (tensor_load_to_lds + s_wait_tensorcnt), double-buffered so the next
//   tile's DMA overlaps the current tile's WMMA work.
// ---------------------------------------------------------------------------

typedef __bf16 bf16;
typedef __attribute__((ext_vector_type(8)))  __bf16 v8bf;
typedef __attribute__((ext_vector_type(16))) __bf16 v16bf;
typedef __attribute__((ext_vector_type(8)))  float  v8f;
typedef __attribute__((ext_vector_type(4)))  unsigned int v4u;
typedef __attribute__((ext_vector_type(8)))  int v8i;
typedef __attribute__((ext_vector_type(4)))  int v4i;

constexpr int Bc   = 4;
constexpr int Sc   = 2048;
constexpr int Hc   = 768;
constexpr int NHc  = 12;
constexpr int HDc  = 64;
constexpr int MLPc = 3072;
constexpr int Mc   = Bc * Sc;   // 8192 tokens
constexpr int QKc  = 2 * Hc;    // 1536: packed Q|K row stride
constexpr int QSc  = 3 * Hc;    // 2304: QKV GEMM N dimension

// ----------------------------- small helpers -------------------------------

__device__ __forceinline__ bf16 f2bf(float f) { return (bf16)f; }

__device__ __forceinline__ v8bf ld8(const bf16* p) { return *(const v8bf*)p; }

__device__ __forceinline__ v16bf combine16(v8bf lo, v8bf hi) {
  v16bf r;
#pragma unroll
  for (int i = 0; i < 8; ++i) { r[i] = lo[i]; r[8 + i] = hi[i]; }
  return r;
}

// 16-lane butterfly reductions via DPP16 (independent per half-wave).
__device__ __forceinline__ float rowmax16(float x) {
  int t;
  t = __builtin_amdgcn_update_dpp(0, __float_as_int(x), 0xB1, 0xF, 0xF, true);
  x = fmaxf(x, __int_as_float(t));
  t = __builtin_amdgcn_update_dpp(0, __float_as_int(x), 0x4E, 0xF, 0xF, true);
  x = fmaxf(x, __int_as_float(t));
  t = __builtin_amdgcn_update_dpp(0, __float_as_int(x), 0x141, 0xF, 0xF, true);
  x = fmaxf(x, __int_as_float(t));
  t = __builtin_amdgcn_update_dpp(0, __float_as_int(x), 0x140, 0xF, 0xF, true);
  x = fmaxf(x, __int_as_float(t));
  return x;
}
__device__ __forceinline__ float rowsum16(float x) {
  int t;
  t = __builtin_amdgcn_update_dpp(0, __float_as_int(x), 0xB1, 0xF, 0xF, true);
  x += __int_as_float(t);
  t = __builtin_amdgcn_update_dpp(0, __float_as_int(x), 0x4E, 0xF, 0xF, true);
  x += __int_as_float(t);
  t = __builtin_amdgcn_update_dpp(0, __float_as_int(x), 0x141, 0xF, 0xF, true);
  x += __int_as_float(t);
  t = __builtin_amdgcn_update_dpp(0, __float_as_int(x), 0x140, 0xF, 0xF, true);
  x += __int_as_float(t);
  return x;
}

// --------------------------- Tensor Data Mover -----------------------------
// 2D tile load: tile_d0 contiguous elements (bf16) per row, tile_d1 rows with
// stride_d0 elements between rows. LDS destination rows are padded:
// pad_amount DWORD-code written every pad_interval DWORD-code (ISA D# §8.4).

__device__ __forceinline__ unsigned lds_off(const void* p) {
  return (unsigned)(unsigned long long)(size_t)p;   // LDS aperture: addr[31:0]
}

__device__ __forceinline__ void tdm_load_2d(
    unsigned ldsaddr, const bf16* gptr,
    unsigned tensor_d0, unsigned tensor_d1,
    unsigned tile_d0, unsigned tile_d1,
    unsigned stride_d0, unsigned padInterval, unsigned padAmount) {
  unsigned long long ga = (unsigned long long)(size_t)gptr;
  v4u g0;
  g0[0] = 1u;                                   // count=1, user descriptor
  g0[1] = ldsaddr;                              // lds_addr (bytes)
  g0[2] = (unsigned)ga;                         // global_addr[31:0]
  g0[3] = (unsigned)(ga >> 32) | (2u << 30);    // global_addr[56:32] | type=2

  unsigned long long u0 =
      (1ull << 16)                               // data_size = 2 bytes
      | (1ull << 20)                             // pad_enable
      | ((unsigned long long)padInterval << 22)  // pad_interval code
      | ((unsigned long long)padAmount << 25)    // pad_amount code
      | ((unsigned long long)(tensor_d0 & 0xFFFFu) << 48);
  unsigned long long u1 =
      (unsigned long long)(tensor_d0 >> 16)
      | ((unsigned long long)tensor_d1 << 16)
      | ((unsigned long long)(tile_d0 & 0xFFFFu) << 48);
  unsigned long long u2 =
      (unsigned long long)(tile_d1 & 0xFFFFu)
      | ((unsigned long long)stride_d0 << 32);   // dim0_stride low 32
  unsigned long long u3 =
      (unsigned long long)(stride_d0 >> 31 >> 1); // dim0_stride high bits

  v8i g1;
  g1[0] = (int)(unsigned)u0; g1[1] = (int)(unsigned)(u0 >> 32);
  g1[2] = (int)(unsigned)u1; g1[3] = (int)(unsigned)(u1 >> 32);
  g1[4] = (int)(unsigned)u2; g1[5] = (int)(unsigned)(u2 >> 32);
  g1[6] = (int)(unsigned)u3; g1[7] = (int)(unsigned)(u3 >> 32);

  v4i z4 = {0, 0, 0, 0};
#if __clang_major__ >= 23
  v8i z8 = {0, 0, 0, 0, 0, 0, 0, 0};
  __builtin_amdgcn_tensor_load_to_lds(g0, g1, z4, z4, z8, 0);
#else
  __builtin_amdgcn_tensor_load_to_lds(g0, g1, z4, z4, 0);
#endif
}

// --------------------- weight conversion (transposed) ----------------------

__global__ __launch_bounds__(256) void pack_qkv_wT_kernel(
    const float* __restrict__ wq, const float* __restrict__ wk,
    const float* __restrict__ wv, bf16* __restrict__ out /* [2304][768] */) {
  int idx = blockIdx.x * 256 + threadIdx.x;   // over 2304*768
  int k = idx % Hc;
  int n = idx / Hc;
  float f = (n < Hc)       ? wq[k * Hc + n]
            : (n < 2 * Hc) ? wk[k * Hc + (n - Hc)]
                           : wv[k * Hc + (n - 2 * Hc)];
  out[idx] = f2bf(f);
}

__global__ __launch_bounds__(256) void pack_qkv_b_kernel(
    const float* __restrict__ bq, const float* __restrict__ bk,
    const float* __restrict__ bv, float* __restrict__ out) {
  int i = blockIdx.x * 256 + threadIdx.x;     // over 2304
  out[i] = (i < Hc) ? bq[i] : (i < 2 * Hc) ? bk[i - Hc] : bv[i - 2 * Hc];
}

__global__ __launch_bounds__(256) void cvtT_bf16_kernel(
    const float* __restrict__ in /* [K][N] */, bf16* __restrict__ out /* [N][K] */,
    int K, int N) {
  int idx = blockIdx.x * 256 + threadIdx.x;   // over K*N
  int k = idx % K;
  int n = idx / K;
  out[idx] = f2bf(in[(size_t)k * N + n]);
}

// ------------------------------- layernorm ---------------------------------

__global__ __launch_bounds__(256) void ln_bf16_kernel(
    const float* __restrict__ x, const float* __restrict__ w,
    const float* __restrict__ b, bf16* __restrict__ out) {
  __shared__ float r1[256], r2[256];
  const int t = threadIdx.x;
  const float* xr = x + (size_t)blockIdx.x * Hc;
  float v0 = xr[t], v1 = xr[t + 256], v2 = xr[t + 512];
  r1[t] = v0 + v1 + v2;
  r2[t] = v0 * v0 + v1 * v1 + v2 * v2;
  __syncthreads();
  for (int off = 128; off > 0; off >>= 1) {
    if (t < off) { r1[t] += r1[t + off]; r2[t] += r2[t + off]; }
    __syncthreads();
  }
  float mean = r1[0] * (1.0f / Hc);
  float var  = r2[0] * (1.0f / Hc) - mean * mean;
  float rst  = rsqrtf(var + 1e-6f);
  bf16* orow = out + (size_t)blockIdx.x * Hc;
  orow[t]       = f2bf((v0 - mean) * rst * w[t]       + b[t]);
  orow[t + 256] = f2bf((v1 - mean) * rst * w[t + 256] + b[t + 256]);
  orow[t + 512] = f2bf((v2 - mean) * rst * w[t + 512] + b[t + 512]);
}

// --------------------------------- GEMM ------------------------------------
// C[M,N] = A[M,K] @ W[K,N] + bias (+res) (+GELU).  A row-major bf16,
// W TRANSPOSED as Wt[n][k] bf16.  Double-buffered TDM staging: wave 0 issues
// tensor_load_to_lds for tile t+1, waits tensorcnt<=2 (tile t complete),
// DMA of t+1 overlaps WMMA on t.

template <bool RES, bool GELU_ACT, bool OUT_BF16, bool QKV>
__global__ __launch_bounds__(256) void gemm_bf16_kernel(
    const bf16* __restrict__ A, const bf16* __restrict__ Wt,
    const float* __restrict__ bias, const float* __restrict__ res,
    float* __restrict__ outF, bf16* __restrict__ outB,
    bf16* __restrict__ vtout, int M, int N, int K) {
  __shared__ __attribute__((aligned(16))) bf16 As[2][128][40];  // pad 8
  __shared__ __attribute__((aligned(16))) bf16 Bt[2][64][40];   // pad 8

  const int tid  = threadIdx.x;
  const int lane = tid & 31;
  const int wave = tid >> 5;
  const int wm   = wave >> 1;     // 0..3
  const int wn   = wave & 1;      // 0..1
  const int half = lane >> 4;
  const int l15  = lane & 15;
  const int mblk = blockIdx.y * 128;
  const int nblk = blockIdx.x * 64;
  const int kbA  = half * 8;      // A frag K base {kb..kb+7, kb+16..kb+23}
  const int kbB  = half * 8;      // B frag K base {kb..kb+7, kb+16..kb+23}
  const bool issuer = (wave == 0);

  auto issueAB = [&](int buf, int k0) {
    tdm_load_2d(lds_off(&As[buf][0][0]), A + (size_t)mblk * K + k0,
                (unsigned)K, (unsigned)M, 32, 128, (unsigned)K, 3, 3);
    tdm_load_2d(lds_off(&Bt[buf][0][0]), Wt + (size_t)nblk * K + k0,
                (unsigned)K, (unsigned)N, 32, 64, (unsigned)K, 3, 3);
  };

  v8f acc[2][2] = {};

  if (issuer) issueAB(0, 0);
  for (int k0 = 0; k0 < K; k0 += 32) {
    const int cur = (k0 >> 5) & 1;
    __syncthreads();                      // buf cur^1 free for overwrite
    if (issuer) {
      if (k0 + 32 < K) {
        issueAB(cur ^ 1, k0 + 32);
        __builtin_amdgcn_s_wait_tensorcnt(2);   // buf cur landed
      } else {
        __builtin_amdgcn_s_wait_tensorcnt(0);
      }
      asm volatile("" ::: "memory");
    }
    __syncthreads();                      // buf cur visible to all waves

    v16bf af[2], bfr[2];
#pragma unroll
    for (int t = 0; t < 2; ++t) {
      int r = wm * 32 + t * 16 + l15;
      af[t] = combine16(ld8(&As[cur][r][kbA]), ld8(&As[cur][r][kbA + 16]));
      int n = wn * 32 + t * 16 + l15;
      bfr[t] = combine16(ld8(&Bt[cur][n][kbB]), ld8(&Bt[cur][n][kbB + 16]));
    }
#pragma unroll
    for (int ti = 0; ti < 2; ++ti)
#pragma unroll
      for (int tj = 0; tj < 2; ++tj)
        acc[ti][tj] = __builtin_amdgcn_wmma_f32_16x16x32_bf16(
            false, af[ti], false, bfr[tj], (short)0, acc[ti][tj], false, false);
  }

  // epilogue
  const bool vregion = QKV && (nblk >= QKc);   // uniform per block
#pragma unroll
  for (int ti = 0; ti < 2; ++ti) {
#pragma unroll
    for (int tj = 0; tj < 2; ++tj) {
      int n = nblk + wn * 32 + tj * 16 + l15;
      float bv = bias[n];
#pragma unroll
      for (int v = 0; v < 8; ++v) {
        int m = mblk + wm * 32 + ti * 16 + v + half * 8;
        float val = acc[ti][tj][v] + bv;
        if (RES)      val += res[(size_t)m * N + n];
        if (GELU_ACT) val = 0.5f * val * (1.0f + erff(val * 0.70710678118f));
        if (QKV) {
          if (!vregion) {
            outB[(size_t)m * QKc + n] = f2bf(val);      // packed Q|K
          } else {                                       // V -> [B][NH][HD][S]
            int nn = n - QKc;
            int bb = m >> 11, ss = m & (Sc - 1);
            int hh = nn >> 6, dd = nn & 63;
            vtout[(((size_t)bb * NHc + hh) * HDc + dd) * Sc + ss] = f2bf(val);
          }
        } else if (OUT_BF16) {
          outB[(size_t)m * N + n] = f2bf(val);
        } else {
          outF[(size_t)m * N + n] = val;
        }
      }
    }
  }
}

// ------------------------------ attention ----------------------------------
// Block = 128 threads (4 waves); each wave owns 16 q rows; block covers 64 q
// rows of one (batch, head). 32 KV positions per step, staged by the TDM
// (double-buffered). K from packed Q|K; V pre-transposed [B][NH][HD][S].

__global__ __launch_bounds__(128) void attn_kernel(
    const bf16* __restrict__ qk, const bf16* __restrict__ vt,
    bf16* __restrict__ ctx) {
  __shared__ __attribute__((aligned(16))) bf16 Kc[2][32][72];    // [kv][d]
  __shared__ __attribute__((aligned(16))) bf16 Vt[2][64][40];    // [d][kv]
  __shared__ __attribute__((aligned(16))) bf16 Pw[4][16][40];    // per-wave P

  const int tid  = threadIdx.x;
  const int lane = tid & 31;
  const int wave = tid >> 5;
  const int half = lane >> 4;
  const int l15  = lane & 15;
  const int kbA  = half * 8;
  const int kbB  = half * 8;
  const int hIdx = blockIdx.y;
  const int bIdx = blockIdx.z;
  const int qblk = blockIdx.x * 64;
  const int tokenBase = bIdx * Sc;
  const bf16* vhead = vt + ((size_t)(bIdx * NHc + hIdx) * HDc) * Sc;
  const bool issuer = (wave == 0);

  auto issueKV = [&](int buf, int s0) {
    tdm_load_2d(lds_off(&Kc[buf][0][0]),
                qk + (size_t)(tokenBase + s0) * QKc + Hc + hIdx * HDc,
                QKc, Mc, 64, 32, QKc, 4, 3);        // 128B rows + 16B pad
    tdm_load_2d(lds_off(&Vt[buf][0][0]), vhead + s0,
                Sc, Bc * NHc * HDc, 32, 64, Sc, 3, 3);  // 64B rows + 16B pad
  };

  // Q fragments (registers for the whole KV sweep)
  const int qrow = qblk + wave * 16 + l15;
  const bf16* qptr = qk + (size_t)(tokenBase + qrow) * QKc + hIdx * HDc;
  v16bf qf[2];
#pragma unroll
  for (int c = 0; c < 2; ++c)
    qf[c] = combine16(ld8(qptr + c * 32 + kbA), ld8(qptr + c * 32 + kbA + 16));

  v8f o[4] = {};
  float mrun[8], lrun[8];
#pragma unroll
  for (int v = 0; v < 8; ++v) { mrun[v] = -1e30f; lrun[v] = 0.0f; }

  if (issuer) issueKV(0, 0);
  for (int s0 = 0; s0 < Sc; s0 += 32) {
    const int cur = (s0 >> 5) & 1;
    __syncthreads();                      // buf cur^1 free for overwrite
    if (issuer) {
      if (s0 + 32 < Sc) {
        issueKV(cur ^ 1, s0 + 32);
        __builtin_amdgcn_s_wait_tensorcnt(2);
      } else {
        __builtin_amdgcn_s_wait_tensorcnt(0);
      }
      asm volatile("" ::: "memory");
    }
    __syncthreads();                      // buf cur ready

    // scores: two 16x16 tiles over 32 kv; accumulate over HD=64
    v8f sc[2];
#pragma unroll
    for (int t = 0; t < 2; ++t) {
      v8f a = {};
      int kv = t * 16 + l15;
#pragma unroll
      for (int c = 0; c < 2; ++c) {
        v16bf kf = combine16(ld8(&Kc[cur][kv][c * 32 + kbB]),
                             ld8(&Kc[cur][kv][c * 32 + kbB + 16]));
        a = __builtin_amdgcn_wmma_f32_16x16x32_bf16(
            false, qf[c], false, kf, (short)0, a, false, false);
      }
      sc[t] = a * 0.125f;  // 1/sqrt(64)
    }

    // online softmax; row = v + 8*half, reduced across 16-lane half via DPP
    float mnew[8];
#pragma unroll
    for (int v = 0; v < 8; ++v)
      mnew[v] = fmaxf(mrun[v], rowmax16(fmaxf(sc[0][v], sc[1][v])));
#pragma unroll
    for (int v = 0; v < 8; ++v) {
      float alpha = __expf(mrun[v] - mnew[v]);
      float p0 = __expf(sc[0][v] - mnew[v]);
      float p1 = __expf(sc[1][v] - mnew[v]);
      lrun[v] = lrun[v] * alpha + rowsum16(p0 + p1);
      mrun[v] = mnew[v];
#pragma unroll
      for (int t = 0; t < 4; ++t) o[t][v] = o[t][v] * alpha;
      int row = v + half * 8;
      Pw[wave][row][l15]      = f2bf(p0);   // D-layout -> LDS
      Pw[wave][row][16 + l15] = f2bf(p1);
    }
    asm volatile("s_wait_dscnt 0" ::: "memory");  // within-wave P relayout

    // P@V: P is A-matrix [16 x 32], V chunk is B-matrix [32 x 64]
    v16bf pf = combine16(ld8(&Pw[wave][l15][kbA]),
                         ld8(&Pw[wave][l15][kbA + 16]));
#pragma unroll
    for (int t = 0; t < 4; ++t) {
      int n = t * 16 + l15;
      v16bf vf = combine16(ld8(&Vt[cur][n][kbB]), ld8(&Vt[cur][n][kbB + 16]));
      o[t] = __builtin_amdgcn_wmma_f32_16x16x32_bf16(
          false, pf, false, vf, (short)0, o[t], false, false);
    }
  }

  // normalize, write ctx bf16 [token][h*64 + d]
#pragma unroll
  for (int v = 0; v < 8; ++v) {
    float inv = 1.0f / lrun[v];
    int token = tokenBase + qblk + wave * 16 + v + half * 8;
#pragma unroll
    for (int t = 0; t < 4; ++t)
      ctx[(size_t)token * Hc + hIdx * HDc + t * 16 + l15] = f2bf(o[t][v] * inv);
  }
}

// ------------------------------- launcher ----------------------------------

extern "C" void kernel_launch(void* const* d_in, const int* in_sizes, int n_in,
                              void* d_out, int out_size, void* d_ws,
                              size_t ws_size, hipStream_t stream) {
  const float* x     = (const float*)d_in[0];
  const float* ln1_w = (const float*)d_in[1];
  const float* ln1_b = (const float*)d_in[2];
  const float* wq    = (const float*)d_in[3];
  const float* bq    = (const float*)d_in[4];
  const float* wk    = (const float*)d_in[5];
  const float* bk    = (const float*)d_in[6];
  const float* wv    = (const float*)d_in[7];
  const float* bv    = (const float*)d_in[8];
  const float* wo    = (const float*)d_in[9];
  const float* bo    = (const float*)d_in[10];
  const float* ln2_w = (const float*)d_in[11];
  const float* ln2_b = (const float*)d_in[12];
  const float* w1    = (const float*)d_in[13];
  const float* b1    = (const float*)d_in[14];
  const float* w2    = (const float*)d_in[15];
  const float* b2    = (const float*)d_in[16];
  float* out = (float*)d_out;

  char* ws = (char*)d_ws;
  size_t off = 0;
  auto alloc = [&](size_t bytes) -> char* {
    char* p = ws + off;
    off += (bytes + 255) & ~(size_t)255;
    return p;
  };
  bf16*  WqkvT = (bf16*) alloc((size_t)QSc * Hc * 2);   // [2304][768]
  float* bqkv  = (float*)alloc((size_t)QSc * 4);
  bf16*  WoT   = (bf16*) alloc((size_t)Hc * Hc * 2);    // [768][768]
  bf16*  W1T   = (bf16*) alloc((size_t)MLPc * Hc * 2);  // [3072][768]
  bf16*  W2T   = (bf16*) alloc((size_t)Hc * MLPc * 2);  // [768][3072]
  bf16*  xn1   = (bf16*) alloc((size_t)Mc * Hc * 2);
  bf16*  qkb   = (bf16*) alloc((size_t)Mc * QKc * 2);   // packed Q|K
  bf16*  vtb   = (bf16*) alloc((size_t)Bc * NHc * HDc * Sc * 2);  // V^T
  bf16*  ctx   = (bf16*) alloc((size_t)Mc * Hc * 2);
  float* x1    = (float*)alloc((size_t)Mc * Hc * 4);
  bf16*  xn2   = (bf16*) alloc((size_t)Mc * Hc * 2);
  bf16*  h1    = (bf16*) alloc((size_t)Mc * MLPc * 2);

  // weight conversion/packing (bf16, transposed [N][K])
  pack_qkv_wT_kernel<<<(QSc * Hc) / 256, 256, 0, stream>>>(wq, wk, wv, WqkvT);
  pack_qkv_b_kernel<<<QSc / 256, 256, 0, stream>>>(bq, bk, bv, bqkv);
  cvtT_bf16_kernel<<<(Hc * Hc) / 256, 256, 0, stream>>>(wo, WoT, Hc, Hc);
  cvtT_bf16_kernel<<<(Hc * MLPc) / 256, 256, 0, stream>>>(w1, W1T, Hc, MLPc);
  cvtT_bf16_kernel<<<(MLPc * Hc) / 256, 256, 0, stream>>>(w2, W2T, MLPc, Hc);

  // LN1
  ln_bf16_kernel<<<Mc, 256, 0, stream>>>(x, ln1_w, ln1_b, xn1);

  // fused QKV projection: Q|K packed + V transposed scatter
  gemm_bf16_kernel<false, false, true, true>
      <<<dim3(QSc / 64, Mc / 128), 256, 0, stream>>>(
          xn1, WqkvT, bqkv, nullptr, nullptr, qkb, vtb, Mc, QSc, Hc);

  // flash attention -> ctx bf16
  attn_kernel<<<dim3(Sc / 64, NHc, Bc), 128, 0, stream>>>(qkb, vtb, ctx);

  // output projection + residual(x) -> x1 f32
  gemm_bf16_kernel<true, false, false, false>
      <<<dim3(Hc / 64, Mc / 128), 256, 0, stream>>>(
          ctx, WoT, bo, x, x1, nullptr, nullptr, Mc, Hc, Hc);

  // LN2
  ln_bf16_kernel<<<Mc, 256, 0, stream>>>(x1, ln2_w, ln2_b, xn2);

  // MLP up + exact GELU -> h1 bf16
  gemm_bf16_kernel<false, true, true, false>
      <<<dim3(MLPc / 64, Mc / 128), 256, 0, stream>>>(
          xn2, W1T, b1, nullptr, nullptr, h1, nullptr, Mc, MLPc, Hc);

  // MLP down + residual(x1) -> d_out f32
  gemm_bf16_kernel<true, false, false, false>
      <<<dim3(Hc / 64, Mc / 128), 256, 0, stream>>>(
          h1, W2T, b2, x1, out, nullptr, nullptr, Mc, Hc, MLPc);
}
